// TokenGeneration_19224273617588
// MI455X (gfx1250) — matrix operand order repaired and checked by reading, not compiled, not run on hardware
//
#include <hip/hip_runtime.h>
#include <math.h>

// ---------------------------------------------------------------------------
// MI455X / gfx1250 implementation.
// GEMMs run through v_wmma_f32_16x16x32_bf16 (bf16 in, f32 accumulate).
// ---------------------------------------------------------------------------

typedef __bf16 bf16_t;
typedef bf16_t v16bf __attribute__((ext_vector_type(16)));
typedef bf16_t v8bf  __attribute__((ext_vector_type(8)));
typedef float  v8f   __attribute__((ext_vector_type(8)));

#define DIM   384
#define BATCH 128
#define NT    64
#define NS    1024
#define H4    1536
#define C2    768
#define NIMG  384          /* BATCH*TOPK */
#define MROWS 6144         /* NIMG*16 */

// ---------------------------------------------------------------- z_max + L2 normalize
__global__ __launch_bounds__(384) void k_zmax_norm(const float* __restrict__ z,
                                                   float* __restrict__ zn) {
  int b = blockIdx.x, c = threadIdx.x;
  float m = -3.4e38f;
  const float* zb = z + (size_t)b * NT * DIM + c;
  for (int t = 0; t < NT; ++t) m = fmaxf(m, zb[(size_t)t * DIM]);
  __shared__ float red[512];
  red[c] = m * m;
  if (c < 128) red[384 + c] = 0.f;
  __syncthreads();
  for (int s = 256; s > 0; s >>= 1) {
    if (c < s) red[c] += red[c + s];
    __syncthreads();
  }
  float inv = rsqrtf(fmaxf(red[0], 1e-24f));
  zn[(size_t)b * DIM + c] = m * inv;
}

// ---------------------------------------------------------------- sim: one wave per x row (bandwidth bound)
__global__ __launch_bounds__(256) void k_sim(const float* __restrict__ x,
                                             const float* __restrict__ zn,
                                             float* __restrict__ sim) {
  int wave = blockIdx.x * 8 + (threadIdx.x >> 5);
  int lane = threadIdx.x & 31;
  int b = wave >> 10, n = wave & 1023;
  const float* xr = x + (size_t)(b * NS + n) * DIM;
  const float* zr = zn + (size_t)b * DIM;
  float dot = 0.f, ss = 0.f;
  for (int i = lane; i < DIM; i += 32) {
    float v = xr[i];
    dot = fmaf(v, zr[i], dot);
    ss  = fmaf(v, v, ss);
  }
  for (int o = 16; o > 0; o >>= 1) {
    dot += __shfl_xor(dot, o, 32);
    ss  += __shfl_xor(ss, o, 32);
  }
  if (lane == 0) sim[(size_t)b * NS + n] = dot * rsqrtf(fmaxf(ss, 1e-24f));
}

// ---------------------------------------------------------------- 2x2 window mean (deterministic)
__global__ __launch_bounds__(256) void k_winmean(const float* __restrict__ sim,
                                                 float* __restrict__ simw) {
  int b = blockIdx.x, w = threadIdx.x;          // 256 windows
  int wy = w >> 4, wx = w & 15;
  const float* s = sim + (size_t)b * NS;
  int n0 = wy * 64 + wx * 2;
  simw[(size_t)b * 256 + w] = 0.25f * (s[n0] + s[n0 + 1] + s[n0 + 32] + s[n0 + 33]);
}

// ---------------------------------------------------------------- top-3 (first-index ties like lax.top_k)
__global__ __launch_bounds__(128) void k_topk(const float* __restrict__ simw,
                                              int* __restrict__ idx) {
  int b = blockIdx.x * blockDim.x + threadIdx.x;
  if (b >= BATCH) return;
  const float* s = simw + (size_t)b * 256;
  int c0 = -1, c1 = -1;
  for (int t = 0; t < 3; ++t) {
    float best = -3.4e38f; int bi = 0;
    for (int i = 0; i < 256; ++i) {
      if (i == c0 || i == c1) continue;
      float v = s[i];
      if (v > best) { best = v; bi = i; }
    }
    idx[b * 3 + t] = bi;
    if (t == 0) c0 = bi; else if (t == 1) c1 = bi;
  }
}

// ---------------------------------------------------------------- tiny gaussian-splat image (C x 16)
__global__ __launch_bounds__(384) void k_gs(const float* __restrict__ xyz,
                                            const float* __restrict__ scal,
                                            const float* __restrict__ rotp,
                                            const float* __restrict__ feat,
                                            const float* __restrict__ opac,
                                            const float* __restrict__ pw,
                                            const float* __restrict__ pb,
                                            float* __restrict__ gsimg) {
  __shared__ float img[16][3];
  int tid = threadIdx.x;
  if (tid < 48) {
    int p = tid / 3, k = tid % 3;
    int py = p >> 2, px = p & 3;
    float acc = 0.f;
    for (int nn = 0; nn < 9; ++nn) {
      float mx = tanhf(xyz[nn * 2 + 0]), my = tanhf(xyz[nn * 2 + 1]);
      float sx = fabsf(scal[nn * 2 + 0] + 0.5f), sy = fabsf(scal[nn * 2 + 1] + 0.5f);
      float rot = (1.f / (1.f + expf(-rotp[nn]))) * 6.28318530717958647692f;
      float cr = cosf(rot), sr = sinf(rot);
      float cov00 = cr * cr * sx * sx + sr * sr * sy * sy;
      float cov01 = sr * cr * (sx * sx - sy * sy);
      float cov11 = sr * sr * sx * sx + cr * cr * sy * sy;
      float det = cov00 * cov11 - cov01 * cov01;
      float ca = cov11 / det, cb = -cov01 / det, cc = cov00 / det;
      float gx = 0.5f * (mx + 1.f) * 4.f - 0.5f;
      float gy = 0.5f * (my + 1.f) * 4.f - 0.5f;
      float dx = gx - (float)px, dy = gy - (float)py;
      float sigma = 0.5f * (ca * dx * dx + cc * dy * dy) + cb * dx * dy;
      float alpha = (sigma >= 0.f) ? opac[nn] * expf(-sigma) : 0.f;
      acc = fmaf(alpha, feat[nn * 3 + k], acc);
    }
    img[p][k] = fminf(fmaxf(acc, 0.f), 1.f);
  }
  __syncthreads();
  int c = tid;
  float w0 = pw[c * 3 + 0], w1 = pw[c * 3 + 1], w2 = pw[c * 3 + 2], bb = pb[c];
  for (int p = 0; p < 16; ++p)
    gsimg[(size_t)c * 16 + p] =
        fmaf(img[p][0], w0, fmaf(img[p][1], w1, fmaf(img[p][2], w2, bb)));
}

// ---------------------------------------------------------------- gather + bilinear 2->4 + LayerNorm -> bf16 A matrix
__global__ __launch_bounds__(384) void k_gather_ln(const float* __restrict__ x,
                                                   const int* __restrict__ idx,
                                                   const float* __restrict__ lng,
                                                   const float* __restrict__ lnb,
                                                   float* __restrict__ xup,
                                                   bf16_t* __restrict__ a1) {
  int row = blockIdx.x;                 // img*16 + p, p = h*4+w
  int img = row >> 4, p = row & 15;
  int h = p >> 2, w = p & 3;
  int b = img / 3, t = img % 3;
  int wi = idx[b * 3 + t];
  int wy = wi >> 4, wx = wi & 15;
  const float tab0[4] = {1.f, 0.75f, 0.25f, 0.f};   // half-pixel linear resize weights
  float wh0 = tab0[h], wh1 = 1.f - wh0;
  float ww0 = tab0[w], ww1 = 1.f - ww0;
  int nbase = wy * 64 + wx * 2;
  const float* xb  = x + (size_t)b * NS * DIM;
  const float* s00 = xb + (size_t)(nbase) * DIM;
  const float* s01 = xb + (size_t)(nbase + 1) * DIM;
  const float* s10 = xb + (size_t)(nbase + 32) * DIM;
  const float* s11 = xb + (size_t)(nbase + 33) * DIM;
  int c = threadIdx.x;
  float v = wh0 * (ww0 * s00[c] + ww1 * s01[c]) + wh1 * (ww0 * s10[c] + ww1 * s11[c]);
  xup[(size_t)row * DIM + c] = v;       // residual, pre-LN
  __shared__ float r1[512], r2[512];
  r1[c] = v; r2[c] = v * v;
  if (c < 128) { r1[384 + c] = 0.f; r2[384 + c] = 0.f; }
  __syncthreads();
  for (int s = 256; s > 0; s >>= 1) {
    if (c < s) { r1[c] += r1[c + s]; r2[c] += r2[c + s]; }
    __syncthreads();
  }
  float mean = r1[0] * (1.f / DIM);
  float var  = r2[0] * (1.f / DIM) - mean * mean;
  float ninv = rsqrtf(var + 1e-5f);
  a1[(size_t)row * DIM + c] = (bf16_t)fmaf((v - mean) * ninv, lng[c], lnb[c]);
}

// ---------------------------------------------------------------- f32 -> bf16 weight conversion
__global__ __launch_bounds__(256) void k_cvt(const float* __restrict__ w_in,
                                             const float* __restrict__ w_mid,
                                             bf16_t* __restrict__ wi_bf,
                                             bf16_t* __restrict__ wm_bf) {
  int i = blockIdx.x * 256 + threadIdx.x;
  const int n1 = H4 * DIM;    // 589824
  const int n2 = C2 * H4;     // 1179648
  if (i < n1) wi_bf[i] = (bf16_t)w_in[i];
  if (i < n2) wm_bf[i] = (bf16_t)w_mid[i];
}

// ---------------------------------------------------------------- WMMA bf16 GEMM: out[M,N] = A[M,K] * W[N,K]^T + bias
// One wave -> 16x64 output slab (4 accumulators). Fragment layouts per CDNA5 ISA 7.12.2.
__global__ __launch_bounds__(256) void k_wmma_gemm(const bf16_t* __restrict__ A,
                                                   const bf16_t* __restrict__ W,
                                                   const float* __restrict__ bias,
                                                   float* __restrict__ out,
                                                   int K, int N) {
  const int wv   = threadIdx.x >> 5;
  const int lane = threadIdx.x & 31;
  const int mBase = (blockIdx.x * 8 + wv) * 16;
  const int nBase = blockIdx.y * 64;
  const int mRow   = lane & 15;
  const int hi     = lane >> 4;     // half-wave select
  const int aShift = hi * 8;        // A: K 0-7 / 8-15 (+16 for upper 8 halves)
  const int bShift = hi * 8;        // B: K 0-7 / 8-15 (+16 for upper 8 halves), same as A

  const bf16_t* aRow = A + (size_t)(mBase + mRow) * K;
  const bf16_t* bRow[4];
#pragma unroll
  for (int j = 0; j < 4; ++j) bRow[j] = W + (size_t)(nBase + j * 16 + mRow) * K;

  v8f acc[4] = {};
  for (int kb = 0; kb < K; kb += 32) {
    v8bf alo = *(const v8bf*)(aRow + kb + aShift);
    v8bf ahi = *(const v8bf*)(aRow + kb + 16 + aShift);
    v16bf a;
#pragma unroll
    for (int e = 0; e < 8; ++e) { a[e] = alo[e]; a[e + 8] = ahi[e]; }
#pragma unroll
    for (int j = 0; j < 4; ++j) {
      v8bf blo = *(const v8bf*)(bRow[j] + kb + bShift);
      v8bf bhi = *(const v8bf*)(bRow[j] + kb + 16 + bShift);
      v16bf bv;
#pragma unroll
      for (int e = 0; e < 8; ++e) { bv[e] = blo[e]; bv[e + 8] = bhi[e]; }
      acc[j] = __builtin_amdgcn_wmma_f32_16x16x32_bf16(
          false, a, false, bv, (short)0, acc[j], false, false);
    }
  }
  // C/D layout: element r -> M = r + 8*hi, N = lane&15
  const int mSt = hi * 8;
  const int nSt = lane & 15;
#pragma unroll
  for (int j = 0; j < 4; ++j) {
    int n = nBase + j * 16 + nSt;
    float bv = bias[n];
#pragma unroll
    for (int r = 0; r < 8; ++r)
      out[(size_t)(mBase + mSt + r) * N + n] = acc[j][r] + bv;
  }
}

// ---------------------------------------------------------------- 4x4 spectral gating (ortho rfft2 * cw -> irfft2), -> bf16
struct Cpx { float r, i; };

__device__ __forceinline__ void dft4(const Cpx* in, Cpx* out, float sgn) {
  Cpx s02 = {in[0].r + in[2].r, in[0].i + in[2].i};
  Cpx d02 = {in[0].r - in[2].r, in[0].i - in[2].i};
  Cpx s13 = {in[1].r + in[3].r, in[1].i + in[3].i};
  Cpx d13 = {in[1].r - in[3].r, in[1].i - in[3].i};
  Cpx jd  = {-sgn * d13.i, sgn * d13.r};        // sgn * i * d13
  out[0] = {s02.r + s13.r, s02.i + s13.i};
  out[1] = {d02.r + jd.r, d02.i + jd.i};
  out[2] = {s02.r - s13.r, s02.i - s13.i};
  out[3] = {d02.r - jd.r, d02.i - jd.i};
}

__global__ __launch_bounds__(256) void k_fftgate(const float* __restrict__ hmid,
                                                 const float* __restrict__ cw,
                                                 bf16_t* __restrict__ hbf) {
  int tid = blockIdx.x * 256 + threadIdx.x;
  if (tid >= NIMG * H4) return;
  int img = tid / H4, o = tid % H4;
  const float* base = hmid + (size_t)img * 16 * H4 + o;
  float hv[4][4];
#pragma unroll
  for (int y = 0; y < 4; ++y)
#pragma unroll
    for (int xx = 0; xx < 4; ++xx) hv[y][xx] = base[(size_t)(y * 4 + xx) * H4];

  Cpx Fx[4][4];
#pragma unroll
  for (int y = 0; y < 4; ++y) {
    Cpx in[4];
    for (int xx = 0; xx < 4; ++xx) in[xx] = {hv[y][xx], 0.f};
    dft4(in, Fx[y], -1.f);
  }
  Cpx F[4][4];
#pragma unroll
  for (int fx = 0; fx < 4; ++fx) {
    Cpx in[4], oc[4];
    for (int y = 0; y < 4; ++y) in[y] = Fx[y][fx];
    dft4(in, oc, -1.f);
    for (int fy = 0; fy < 4; ++fy) F[fy][fx] = oc[fy];
  }
  // multiply by cw (Hermitian extension for fx==3)
  Cpx G[4][4];
#pragma unroll
  for (int fy = 0; fy < 4; ++fy)
#pragma unroll
    for (int fx = 0; fx < 4; ++fx) {
      float wr, wi;
      if (fx < 3) {
        size_t wo = ((size_t)(fy * 3 + fx) * H4 + o) * 2;
        wr = cw[wo]; wi = cw[wo + 1];
      } else {
        int fy2 = (4 - fy) & 3;
        size_t wo = ((size_t)(fy2 * 3 + 1) * H4 + o) * 2;
        wr = cw[wo]; wi = -cw[wo + 1];
      }
      Cpx f = F[fy][fx];
      G[fy][fx] = {f.r * wr - f.i * wi, f.r * wi + f.i * wr};
    }
  // inverse
  Cpx Ty[4][4];
#pragma unroll
  for (int fx = 0; fx < 4; ++fx) {
    Cpx in[4], oc[4];
    for (int fy = 0; fy < 4; ++fy) in[fy] = G[fy][fx];
    dft4(in, oc, 1.f);
    for (int y = 0; y < 4; ++y) Ty[y][fx] = oc[y];
  }
  bf16_t* ob = hbf + (size_t)img * 16 * H4 + o;
#pragma unroll
  for (int y = 0; y < 4; ++y) {
    Cpx oc[4];
    dft4(Ty[y], oc, 1.f);
    for (int xx = 0; xx < 4; ++xx)
      ob[(size_t)(y * 4 + xx) * H4] = (bf16_t)(oc[xx].r * (1.f / 16.f));
  }
}

// ---------------------------------------------------------------- gelu(x1)*x2 + residual + gs, final layout write
__global__ __launch_bounds__(384) void k_final(const float* __restrict__ out2,
                                               const float* __restrict__ xup,
                                               const float* __restrict__ gsimg,
                                               float* __restrict__ out) {
  int row = blockIdx.x, c = threadIdx.x;
  int p = row & 15;
  float x1 = out2[(size_t)row * C2 + c];
  float x2 = out2[(size_t)row * C2 + DIM + c];
  float g = 0.5f * x1 * (1.f + erff(x1 * 0.70710678118654752f));
  out[(size_t)row * DIM + c] =
      fmaf(g, x2, xup[(size_t)row * DIM + c] + gsimg[(size_t)c * 16 + p]);
}

// ---------------------------------------------------------------- launch
extern "C" void kernel_launch(void* const* d_in, const int* in_sizes, int n_in,
                              void* d_out, int out_size, void* d_ws, size_t ws_size,
                              hipStream_t stream) {
  (void)in_sizes; (void)n_in; (void)out_size; (void)ws_size;
  const float* z           = (const float*)d_in[0];
  const float* x           = (const float*)d_in[1];
  const float* ln_g        = (const float*)d_in[2];
  const float* ln_b        = (const float*)d_in[3];
  const float* w_in        = (const float*)d_in[4];
  const float* b_in        = (const float*)d_in[5];
  const float* cw          = (const float*)d_in[6];
  const float* w_mid       = (const float*)d_in[7];
  const float* b_mid       = (const float*)d_in[8];
  const float* gs_xyz      = (const float*)d_in[9];
  const float* gs_scaling  = (const float*)d_in[10];
  const float* gs_rotation = (const float*)d_in[11];
  const float* gs_features = (const float*)d_in[12];
  const float* gs_opacity  = (const float*)d_in[13];
  const float* proj_w      = (const float*)d_in[14];
  const float* proj_b      = (const float*)d_in[15];
  float* out = (float*)d_out;

  // workspace layout (f32 region then bf16 region, all 64B aligned; ~94 MB total)
  float* wsf = (float*)d_ws;
  char*  wsb = (char*)d_ws;
  float* zn    = wsf + 0;            // 49152
  float* sim   = wsf + 49152;        // 131072
  float* simw  = wsf + 180224;       // 32768
  int*   idx   = (int*)(wsf + 212992); // 384 ints
  float* xup   = wsf + 213376;       // 2359296
  float* gsimg = wsf + 2572672;      // 6144
  float* hmid  = wsf + 2578816;      // 9437184
  float* out2  = wsf + 12016000;     // 4718592  -> f32 end = 16734592 floats
  bf16_t* a1    = (bf16_t*)(wsb + 66938368); // 2359296 bf16
  bf16_t* wi_bf = (bf16_t*)(wsb + 71656960); // 589824
  bf16_t* wm_bf = (bf16_t*)(wsb + 72836608); // 1179648
  bf16_t* hbf   = (bf16_t*)(wsb + 75195904); // 9437184 -> end 94070272 B

  k_gs<<<1, 384, 0, stream>>>(gs_xyz, gs_scaling, gs_rotation, gs_features,
                              gs_opacity, proj_w, proj_b, gsimg);
  k_zmax_norm<<<BATCH, 384, 0, stream>>>(z, zn);
  k_sim<<<(BATCH * NS) / 8, 256, 0, stream>>>(x, zn, sim);
  k_winmean<<<BATCH, 256, 0, stream>>>(sim, simw);
  k_topk<<<1, 128, 0, stream>>>(simw, idx);
  k_gather_ln<<<MROWS, 384, 0, stream>>>(x, idx, ln_g, ln_b, xup, a1);
  k_cvt<<<(C2 * H4 + 255) / 256, 256, 0, stream>>>(w_in, w_mid, wi_bf, wm_bf);
  {
    dim3 g1(MROWS / 128, H4 / 64);   // (48, 24)
    k_wmma_gemm<<<g1, 256, 0, stream>>>(a1, wi_bf, b_in, hmid, DIM, H4);
  }
  k_fftgate<<<(NIMG * H4 + 255) / 256, 256, 0, stream>>>(hmid, cw, hbf);
  {
    dim3 g2(MROWS / 128, C2 / 64);   // (48, 12)
    k_wmma_gemm<<<g2, 256, 0, stream>>>(hbf, wm_bf, b_mid, out2, H4, C2);
  }
  k_final<<<MROWS, 384, 0, stream>>>(out2, xup, gsimg, out);
}
